// TemporalSelfAttention_66417374265380
// MI455X (gfx1250) — matrix operand order repaired and checked by reading, not compiled, not run on hardware
//
#include <hip/hip_runtime.h>
#include <hip/hip_bf16.h>

// ---------------------------------------------------------------------------
// TemporalSelfAttention for MI455X (gfx1250, wave32, WMMA).
// All heavy GEMMs use v_wmma_f32_16x16x32_bf16 (bf16 inputs, f32 accum).
// Workspace layout (floats unless noted), S = B*T*N*D = 18,874,368 elems:
//   [0,S)    Q     [S,2S)  K     [2S,3S) V
//   [3S,4S)  tc_x  [4S,5S) t_x
//   then: a_bf16 (N*N u16, 2MB), Wbf[8] (8*96*96 u16)
// Total ~380MB of d_ws.
// ---------------------------------------------------------------------------

typedef __attribute__((ext_vector_type(16))) __bf16        v16bf;
typedef __attribute__((ext_vector_type(8)))  float         v8f;
typedef __attribute__((ext_vector_type(8)))  unsigned int  v8u;

namespace {
constexpr int Bc = 8, Tc = 24, Nc = 1024, Dc = 96, Hc = 6, HDc = 16;
constexpr float SCALEc = 0.25f;      // HD^-0.5 = 1/4
constexpr float EPSc   = 1e-6f;
constexpr float ALPHAc = 0.05f;
}

__device__ __forceinline__ unsigned short f32_to_bf16(float f) {
  unsigned int u = __float_as_uint(f);
  u += 0x7FFFu + ((u >> 16) & 1u);   // round-to-nearest-even
  return (unsigned short)(u >> 16);
}

// --------------------------- prep kernels ----------------------------------

__global__ void adj_norm_kernel(const float* __restrict__ adj,
                                unsigned short* __restrict__ abf) {
  __shared__ float red[256];
  const int n = blockIdx.x, tid = threadIdx.x;
  float part = 0.f;
  for (int k = tid; k < Nc; k += 256) part += adj[n * Nc + k];
  red[tid] = part;
  __syncthreads();
  for (int s = 128; s > 0; s >>= 1) {
    if (tid < s) red[tid] += red[tid + s];
    __syncthreads();
  }
  const float inv = 1.f / red[0];
  for (int k = tid; k < Nc; k += 256)
    abf[n * Nc + k] = f32_to_bf16(adj[n * Nc + k] * inv);
}

__global__ void cvt_bf16_kernel(const float* __restrict__ src,
                                unsigned short* __restrict__ dst, int n) {
  const int i = blockIdx.x * 256 + threadIdx.x;
  if (i < n) dst[i] = f32_to_bf16(src[i]);
}

// --------------------- generic [M,96] x [96,96]^T GEMM ---------------------
// Y = X @ W^T.  W given in torch Linear layout [E=96, D=96] as bf16; the
// WMMA B fragment wants B[k][n] = W[n][k], so we index W rows by n directly.
// Block = 256 threads (8 waves), 128 rows of X per block, LDS-staged bf16 A.

__global__ void gemm96_kernel(const float* __restrict__ X,
                              const unsigned short* __restrict__ Wbf,
                              float* __restrict__ Y) {
  __shared__ unsigned int xs32[128 * 48];              // 128 rows x 96 bf16
  const int row0 = blockIdx.x * 128;
  const int tid = threadIdx.x;

  unsigned short* xs16 = (unsigned short*)xs32;
  for (int i = tid; i < 128 * 96; i += 256) {
    const int r = i / 96, c = i % 96;
    xs16[i] = f32_to_bf16(X[(size_t)(row0 + r) * 96 + c]);
  }
  __syncthreads();

  const int wave = tid >> 5, lane = tid & 31;
  const int half = lane >> 4, l16 = lane & 15;
  const unsigned int* w32 = (const unsigned int*)Wbf;  // [96][48] u32 pairs

  v8f acc[6] = {};
#pragma unroll
  for (int ks = 0; ks < 3; ++ks) {
    v8u araw;
#pragma unroll
    for (int v = 0; v < 8; ++v) {
      const int k0 = ks * 32 + half * 16 + 2 * v;
      araw[v] = xs32[(wave * 16 + l16) * 48 + (k0 >> 1)];
    }
    const v16bf af = __builtin_bit_cast(v16bf, araw);
#pragma unroll
    for (int nt = 0; nt < 6; ++nt) {
      v8u braw;
#pragma unroll
      for (int v = 0; v < 8; ++v) {
        const int k0 = ks * 32 + half * 16 + 2 * v;
        braw[v] = w32[(nt * 16 + l16) * 48 + (k0 >> 1)];
      }
      const v16bf bf = __builtin_bit_cast(v16bf, braw);
      acc[nt] = __builtin_amdgcn_wmma_f32_16x16x32_bf16(
          false, af, false, bf, (short)0, acc[nt], false, false);
    }
  }
#pragma unroll
  for (int nt = 0; nt < 6; ++nt)
#pragma unroll
    for (int v = 0; v < 8; ++v) {
      const int r = wave * 16 + v + 8 * half;
      const int c = nt * 16 + l16;
      Y[(size_t)(row0 + r) * 96 + c] = acc[nt][v];
    }
}

// ------------- fused attention + residual + RMSNorm per (b,n) --------------
// Block = 192 threads (6 waves).  wave = head h, lane = query index t (<24).

__global__ void attn_rms_kernel(const float* __restrict__ Q,
                                const float* __restrict__ K,
                                const float* __restrict__ V,
                                const float* __restrict__ RES,
                                const float* __restrict__ g,
                                float* __restrict__ OUT,
                                const int* __restrict__ causal_flag) {
  __shared__ float ksm[Tc * Dc], vsm[Tc * Dc], osm[Tc * Dc];
  const int b = blockIdx.x / Nc;
  const int n = blockIdx.x % Nc;
  const int tid = threadIdx.x;

  for (int i = tid; i < Tc * Dc; i += 192) {
    const int t = i / Dc, d = i % Dc;
    const size_t gi = (((size_t)b * Tc + t) * Nc + n) * Dc + d;
    ksm[i] = K[gi];
    vsm[i] = V[gi];
  }
  __syncthreads();

  const int h = tid / 32;        // head
  const int t = tid % 32;        // query index (valid < 24)
  const bool active = (t < Tc);
  const bool causal = (*causal_flag) != 0;

  float q[HDc];
  if (active) {
    const size_t gi = (((size_t)b * Tc + t) * Nc + n) * Dc + h * HDc;
#pragma unroll
    for (int d = 0; d < HDc; ++d) q[d] = Q[gi + d];
  } else {
#pragma unroll
    for (int d = 0; d < HDc; ++d) q[d] = 0.f;
  }

  float s[Tc];
  float mx = -1e30f;
#pragma unroll
  for (int tk = 0; tk < Tc; ++tk) {
    float dot = 0.f;
#pragma unroll
    for (int d = 0; d < HDc; ++d) dot += q[d] * ksm[tk * Dc + h * HDc + d];
    dot *= SCALEc;
    if (causal && tk > t) dot = -1e9f;
    s[tk] = dot;
    mx = fmaxf(mx, dot);
  }
  float sum = 0.f;
#pragma unroll
  for (int tk = 0; tk < Tc; ++tk) { s[tk] = __expf(s[tk] - mx); sum += s[tk]; }
  const float inv = 1.f / sum;

  float o[HDc] = {};
#pragma unroll
  for (int tk = 0; tk < Tc; ++tk)
#pragma unroll
    for (int d = 0; d < HDc; ++d) o[d] += s[tk] * vsm[tk * Dc + h * HDc + d];

  if (active)
#pragma unroll
    for (int d = 0; d < HDc; ++d) osm[t * Dc + h * HDc + d] = o[d] * inv;
  __syncthreads();

  // residual + RMSNorm, one row per wave iteration; lane covers 3 columns
  for (int r = h; r < Tc; r += Hc) {
    float vals[3];
    float part = 0.f;
#pragma unroll
    for (int j = 0; j < 3; ++j) {
      const int c = t + 32 * j;
      const float xv = osm[r * Dc + c] +
                       RES[(((size_t)b * Tc + r) * Nc + n) * Dc + c];
      vals[j] = xv;
      part += xv * xv;
    }
    for (int off = 16; off > 0; off >>= 1) part += __shfl_xor(part, off, 32);
    const float sc = rsqrtf(part * (1.f / 96.f) + EPSc);
#pragma unroll
    for (int j = 0; j < 3; ++j) {
      const int c = t + 32 * j;
      OUT[(((size_t)b * Tc + r) * Nc + n) * Dc + c] = g[c] * vals[j] * sc;
    }
  }
}

// ------------------- fused GCN + fc1 + blend + mlp -------------------------
// One block per ((b,t), 128-row tile).  K=1024 in two 512-chunks staged to
// LDS transposed as bf16.  Dynamic LDS: 96*512*2 + 128*96*2 = 122,880 B.

__global__ void gcn_fused_kernel(const unsigned short* __restrict__ Abf,
                                 const float* __restrict__ TX,
                                 const unsigned short* __restrict__ Wfc1,
                                 const unsigned short* __restrict__ Wmlp,
                                 const float* __restrict__ bias,
                                 float* __restrict__ HO) {
  extern __shared__ char smem[];
  unsigned short* Xt = (unsigned short*)smem;            // [96][512] bf16
  unsigned short* fb = (unsigned short*)(smem + 96 * 512 * 2);  // [128][96]

  const int bt   = blockIdx.x >> 3;       // (b*T + t)
  const int row0 = (blockIdx.x & 7) * 128;
  const float* X = TX + (size_t)bt * Nc * Dc;
  const int tid = threadIdx.x;
  const int wave = tid >> 5, lane = tid & 31;
  const int half = lane >> 4, l16 = lane & 15;
  const int arow = row0 + wave * 16 + l16;

  const unsigned int* A32  = (const unsigned int*)Abf;   // [1024][512] u32
  const unsigned int* Xt32 = (const unsigned int*)Xt;    // [96][256] u32
  const unsigned int* fb32 = (const unsigned int*)fb;    // [128][48] u32

  // ---- AGG = Â(tile rows) @ X,  K = 1024 ----
  v8f acc[6] = {};
  for (int chunk = 0; chunk < 2; ++chunk) {
    const int kc = chunk * 512;
    __syncthreads();
    for (int i = tid; i < 512 * 96; i += 256) {
      const int k = i / 96, d = i % 96;
      Xt[d * 512 + k] = f32_to_bf16(X[(size_t)(kc + k) * 96 + d]);
    }
    __syncthreads();
    for (int ks = 0; ks < 16; ++ks) {
      v8u araw;
#pragma unroll
      for (int v = 0; v < 8; ++v) {
        const int k0 = ks * 32 + half * 16 + 2 * v;
        araw[v] = A32[(size_t)arow * 512 + ((kc + k0) >> 1)];
      }
      const v16bf af = __builtin_bit_cast(v16bf, araw);
#pragma unroll
      for (int nt = 0; nt < 6; ++nt) {
        v8u braw;
#pragma unroll
        for (int v = 0; v < 8; ++v) {
          const int k0 = ks * 32 + half * 16 + 2 * v;
          braw[v] = Xt32[(nt * 16 + l16) * 256 + (k0 >> 1)];
        }
        const v16bf bf = __builtin_bit_cast(v16bf, braw);
        acc[nt] = __builtin_amdgcn_wmma_f32_16x16x32_bf16(
            false, af, false, bf, (short)0, acc[nt], false, false);
      }
    }
  }

  // stage AGG (bf16) for GEMM2
  __syncthreads();
#pragma unroll
  for (int nt = 0; nt < 6; ++nt)
#pragma unroll
    for (int v = 0; v < 8; ++v) {
      const int r = wave * 16 + v + 8 * half;
      const int c = nt * 16 + l16;
      fb[r * 96 + c] = f32_to_bf16(acc[nt][v]);
    }
  __syncthreads();

  // ---- GEMM2: Y1 = AGG @ W_fc1^T ----
  const unsigned int* w1 = (const unsigned int*)Wfc1;    // [96][48]
  v8f acc2[6] = {};
#pragma unroll
  for (int ks = 0; ks < 3; ++ks) {
    v8u araw;
#pragma unroll
    for (int v = 0; v < 8; ++v) {
      const int k0 = ks * 32 + half * 16 + 2 * v;
      araw[v] = fb32[(wave * 16 + l16) * 48 + (k0 >> 1)];
    }
    const v16bf af = __builtin_bit_cast(v16bf, araw);
#pragma unroll
    for (int nt = 0; nt < 6; ++nt) {
      v8u braw;
#pragma unroll
      for (int v = 0; v < 8; ++v) {
        const int k0 = ks * 32 + half * 16 + 2 * v;
        braw[v] = w1[(nt * 16 + l16) * 48 + (k0 >> 1)];
      }
      const v16bf bf = __builtin_bit_cast(v16bf, braw);
      acc2[nt] = __builtin_amdgcn_wmma_f32_16x16x32_bf16(
          false, af, false, bf, (short)0, acc2[nt], false, false);
    }
  }

  // ---- out = alpha*t_x + (1-alpha)*Y1, staged bf16 for GEMM3 ----
  __syncthreads();
#pragma unroll
  for (int nt = 0; nt < 6; ++nt)
#pragma unroll
    for (int v = 0; v < 8; ++v) {
      const int r = wave * 16 + v + 8 * half;
      const int c = nt * 16 + l16;
      const float txv = X[(size_t)(row0 + r) * 96 + c];
      fb[r * 96 + c] =
          f32_to_bf16(ALPHAc * txv + (1.f - ALPHAc) * acc2[nt][v]);
    }
  __syncthreads();

  // ---- GEMM3: ho = out @ W_mlp^T + b ----
  const unsigned int* w2 = (const unsigned int*)Wmlp;    // [96][48]
  v8f acc3[6] = {};
#pragma unroll
  for (int ks = 0; ks < 3; ++ks) {
    v8u araw;
#pragma unroll
    for (int v = 0; v < 8; ++v) {
      const int k0 = ks * 32 + half * 16 + 2 * v;
      araw[v] = fb32[(wave * 16 + l16) * 48 + (k0 >> 1)];
    }
    const v16bf af = __builtin_bit_cast(v16bf, araw);
#pragma unroll
    for (int nt = 0; nt < 6; ++nt) {
      v8u braw;
#pragma unroll
      for (int v = 0; v < 8; ++v) {
        const int k0 = ks * 32 + half * 16 + 2 * v;
        braw[v] = w2[(nt * 16 + l16) * 48 + (k0 >> 1)];
      }
      const v16bf bf = __builtin_bit_cast(v16bf, braw);
      acc3[nt] = __builtin_amdgcn_wmma_f32_16x16x32_bf16(
          false, af, false, bf, (short)0, acc3[nt], false, false);
    }
  }
#pragma unroll
  for (int nt = 0; nt < 6; ++nt)
#pragma unroll
    for (int v = 0; v < 8; ++v) {
      const int r = wave * 16 + v + 8 * half;
      const int c = nt * 16 + l16;
      HO[(size_t)bt * Nc * Dc + (size_t)(row0 + r) * 96 + c] =
          acc3[nt][v] + bias[c];
    }
}

// ---------------------------------------------------------------------------

extern "C" void kernel_launch(void* const* d_in, const int* in_sizes, int n_in,
                              void* d_out, int out_size, void* d_ws,
                              size_t ws_size, hipStream_t stream) {
  (void)in_sizes; (void)n_in; (void)out_size; (void)ws_size;

  const float* x_q   = (const float*)d_in[0];
  // d_in[1] = x_k, d_in[2] = x_v : unused by the reference
  const float* TH    = (const float*)d_in[3];
  const float* TP    = (const float*)d_in[4];
  const float* adj   = (const float*)d_in[5];
  const float* Wq_tc = (const float*)d_in[6];
  const float* Wk_tc = (const float*)d_in[7];
  const float* Wv_tc = (const float*)d_in[8];
  const float* Wq_t  = (const float*)d_in[9];
  const float* Wk_t  = (const float*)d_in[10];
  const float* Wv_t  = (const float*)d_in[11];
  const float* g1    = (const float*)d_in[12];
  const float* g2    = (const float*)d_in[13];
  const float* W_fc1 = (const float*)d_in[14];
  const float* W_mlp = (const float*)d_in[15];
  const float* b_mlp = (const float*)d_in[16];
  const int*   trg   = (const int*)d_in[17];

  const size_t S = (size_t)Bc * Tc * Nc * Dc;  // 18,874,368
  float* ws  = (float*)d_ws;
  float* Q   = ws;
  float* K   = ws + S;
  float* V   = ws + 2 * S;
  float* tcx = ws + 3 * S;
  float* tx  = ws + 4 * S;
  unsigned short* abf = (unsigned short*)(ws + 5 * S);
  unsigned short* wbf = abf + (size_t)Nc * Nc;  // 8 x 9216 bf16 weights
  float* out = (float*)d_out;

  const int WE = Dc * Dc;  // 9216
  const int M_TILES = (int)(S / (Dc * 128));   // 1536 row-tile blocks

  // prep
  adj_norm_kernel<<<Nc, 256, 0, stream>>>(adj, abf);
  cvt_bf16_kernel<<<36, 256, 0, stream>>>(Wq_tc, wbf + 0 * WE, WE);
  cvt_bf16_kernel<<<36, 256, 0, stream>>>(Wk_tc, wbf + 1 * WE, WE);
  cvt_bf16_kernel<<<36, 256, 0, stream>>>(Wv_tc, wbf + 2 * WE, WE);
  cvt_bf16_kernel<<<36, 256, 0, stream>>>(Wq_t,  wbf + 3 * WE, WE);
  cvt_bf16_kernel<<<36, 256, 0, stream>>>(Wk_t,  wbf + 4 * WE, WE);
  cvt_bf16_kernel<<<36, 256, 0, stream>>>(Wv_t,  wbf + 5 * WE, WE);
  cvt_bf16_kernel<<<36, 256, 0, stream>>>(W_fc1, wbf + 6 * WE, WE);
  cvt_bf16_kernel<<<36, 256, 0, stream>>>(W_mlp, wbf + 7 * WE, WE);

  // attention 1: q from TP, k from TH, v from x_q; residual x_q, norm g1
  gemm96_kernel<<<M_TILES, 256, 0, stream>>>(TP,  wbf + 0 * WE, Q);
  gemm96_kernel<<<M_TILES, 256, 0, stream>>>(TH,  wbf + 1 * WE, K);
  gemm96_kernel<<<M_TILES, 256, 0, stream>>>(x_q, wbf + 2 * WE, V);
  attn_rms_kernel<<<Bc * Nc, 192, 0, stream>>>(Q, K, V, x_q, g1, tcx, trg);

  // attention 2: self-attention on tc_x; residual tc_x, norm g2
  gemm96_kernel<<<M_TILES, 256, 0, stream>>>(tcx, wbf + 3 * WE, Q);
  gemm96_kernel<<<M_TILES, 256, 0, stream>>>(tcx, wbf + 4 * WE, K);
  gemm96_kernel<<<M_TILES, 256, 0, stream>>>(tcx, wbf + 5 * WE, V);
  attn_rms_kernel<<<Bc * Nc, 192, 0, stream>>>(Q, K, V, tcx, g2, tx, trg);

  // fused GCN + fc1 + blend + mlp
  const int SMEM = 96 * 512 * 2 + 128 * 96 * 2;  // 122,880 B dynamic LDS
  (void)hipFuncSetAttribute((const void*)gcn_fused_kernel,
                            hipFuncAttributeMaxDynamicSharedMemorySize, SMEM);
  gcn_fused_kernel<<<Bc * Tc * 8, 256, SMEM, stream>>>(
      abf, tx, wbf + 6 * WE, wbf + 7 * WE, b_mlp, out);
}
